// ToyGraphActorCriticModel_68685116998264
// MI455X (gfx1250) — matrix operand order repaired and checked by reading, not compiled, not run on hardware
//
#include <hip/hip_runtime.h>
#include <hip/hip_bf16.h>
#include <math.h>

// ---------------------------------------------------------------------------
// MPNN (NNConv + GRU x6 + Set2Set(1) + Linear) for gfx1250 / MI455X, wave32.
// Edge GEMM:  msg = U @ V,  U[e, k*64+d] = he[e,k] * h[src[e], d]  (built in
// registers), V = reshaped We2 (+be2 block), K = 4160 = 130 slabs of 32,
// bf16 WMMA 16x16x32, fp32 accumulate.  B streamed into LDS by the Tensor
// Data Mover (tensor_load_to_lds, TENSORcnt), double-buffered.
// Key identity: within slab s, kidx = s>>1 (one he value per lane per slab),
// didx = (s&1)*32 + kk (t values hoisted to registers for both parities).
// ---------------------------------------------------------------------------

#define DDIM    64
#define NNODES  4096
#define NEDGES  16384
#define BGRAPH  16
#define NSLOPE  0.01f
#define KEDGE   4160          // 65*64 (bias folded as he==1 block)
#define SLABS_E 130           // KEDGE/32
#define NT_EDGE 4             // 64 cols / 16
#define CH      10            // K-slabs per TDM chunk (even!)
#define NCH     13            // SLABS_E / CH

typedef __attribute__((ext_vector_type(16))) __bf16 v16bf;
typedef __attribute__((ext_vector_type(8)))  float  v8f;
typedef __attribute__((ext_vector_type(4)))  unsigned u32x4;
typedef __attribute__((ext_vector_type(8)))  int      i32x8;

__device__ __forceinline__ v8f wmma_bf16(v16bf a, v16bf b, v8f c) {
  return __builtin_amdgcn_wmma_f32_16x16x32_bf16(false, a, false, b,
                                                 (short)0, c, false, false);
}

__device__ __forceinline__ float lrelu_f(float x) { return x >= 0.f ? x : NSLOPE * x; }
__device__ __forceinline__ float sigm_f (float x) { return 1.f / (1.f + __expf(-x)); }
__device__ __forceinline__ void  atomAddF(float* p, float v) {
  __hip_atomic_fetch_add(p, v, __ATOMIC_RELAXED, __HIP_MEMORY_SCOPE_AGENT);
}

// Generic LDS pointer -> LDS byte offset (ISA 10.2: LDS aperture uses addr[31:0]).
__device__ __forceinline__ unsigned lds_off(const void* p) {
  return (unsigned)(unsigned long long)p;
}

// Tensor Data Mover: 2D tile (rows x row_words DWORDs) global -> LDS.
// D# group0/group1 built per CDNA5 ISA sect. 8.3/8.4; issued once per wave.
__device__ __forceinline__ void tdm_load_2d(unsigned ldsaddr, const void* g,
                                            unsigned row_words, unsigned rows,
                                            unsigned stride_words) {
  unsigned long long ga = (unsigned long long)g;
  u32x4 g0;
  g0[0] = 1u;                                            // count=1, user D#
  g0[1] = ldsaddr;                                       // lds_addr (bytes)
  g0[2] = (unsigned)ga;                                  // global_addr[31:0]
  g0[3] = (unsigned)((ga >> 32) & 0x01FFFFFFu) | (2u << 30); // addr[56:32]|type=2
  i32x8 g1;
  g1[0] = (int)(2u << 16);                               // data_size = 4B
  g1[1] = (int)((row_words & 0xFFFFu) << 16);            // tensor_dim0[15:0]
  g1[2] = (int)(((row_words >> 16) & 0xFFFFu)            // tensor_dim0[31:16]
              | ((rows & 0xFFFFu) << 16));               // tensor_dim1[15:0]
  g1[3] = (int)((row_words & 0xFFFFu) << 16);            // tile_dim0
  g1[4] = (int)(rows & 0xFFFFu);                         // tile_dim1 (tile_dim2=0)
  g1[5] = (int)stride_words;                             // tensor_dim0_stride lo
  g1[6] = 0;                                             // stride hi / dim1_stride lo
  g1[7] = 0;
  asm volatile("tensor_load_to_lds %0, %1" :: "s"(g0), "s"(g1) : "memory");
}

// A-fragment K index inside a 32-slab (16-bit A 16x32, ISA 7.12.2):
//   lane half=0: j0..7 -> K0..7,  j8..15 -> K16..23
//   lane half=1: j0..7 -> K8..15, j8..15 -> K24..31
__device__ __forceinline__ int a_kk(int j, int half) {
  return j + ((j >= 8) ? 8 : 0) + half * 8;
}

// ------------------------------- utility -----------------------------------
__global__ void k_zero(float* __restrict__ p, int n) {
  int i = blockIdx.x * blockDim.x + threadIdx.x;
  if (i < n) p[i] = 0.f;
}

// ------------------------- initial embeddings -------------------------------
__global__ void k_node_init(const float* __restrict__ x, const float* __restrict__ W0,
                            const float* __restrict__ b0, float* __restrict__ h) {
  int i = blockIdx.x * blockDim.x + threadIdx.x;        // NNODES*64
  int nd = i >> 6, f = i & 63;
  float a = x[nd*3+0]*W0[f] + x[nd*3+1]*W0[64+f] + x[nd*3+2]*W0[128+f] + b0[f];
  h[i] = lrelu_f(a);
}

__global__ void k_edge_mlp(const float* __restrict__ ea, const float* __restrict__ We1,
                           const float* __restrict__ be1, float* __restrict__ he) {
  int i = blockIdx.x * blockDim.x + threadIdx.x;        // NEDGES*64
  int e = i >> 6, f = i & 63;
  float a = ea[e*4+0]*We1[f] + ea[e*4+1]*We1[64+f] + ea[e*4+2]*We1[128+f]
          + ea[e*4+3]*We1[192+f] + be1[f];
  he[i] = lrelu_f(a);
}

__global__ void k_deg(const int* __restrict__ dst, float* __restrict__ deg) {
  int e = blockIdx.x * blockDim.x + threadIdx.x;
  if (e < NEDGES) atomAddF(&deg[dst[e]], 1.0f);
}

__global__ void k_invdeg(const float* __restrict__ deg, float* __restrict__ inv) {
  int i = blockIdx.x * blockDim.x + threadIdx.x;
  if (i < NNODES) inv[i] = 1.0f / fmaxf(deg[i], 1.0f);
}

// ---------------------- B-fragment packing (bf16) ---------------------------
// B 32x16 bf16 fragment (ISA 7.12.2): lane -> N = lane%16, K = a_kk(j, lane/16) (same order as A).
// Layout in memory: [ntile][slab][lane][16] so each lane loads 32 contiguous B.

// Edge V matrix: V[k*64+d][f] = We2[k][d*64+f] for k<64, = be2[d*64+f] at k==64.
__global__ void k_pack_edgeB(const float* __restrict__ We2, const float* __restrict__ be2,
                             __bf16* __restrict__ Bf) {
  int idx = blockIdx.x * blockDim.x + threadIdx.x;      // NT_EDGE*SLABS_E*32
  if (idx >= NT_EDGE * SLABS_E * 32) return;
  int lane = idx & 31;
  int s    = (idx >> 5) % SLABS_E;
  int nt   = idx / (SLABS_E * 32);
  int n = lane & 15, half = lane >> 4;
  int f = nt * 16 + n;
  #pragma unroll
  for (int j = 0; j < 16; ++j) {
    int kg = s*32 + a_kk(j, half);
    int kidx = kg >> 6, didx = kg & 63;
    float v = (kidx < 64) ? We2[(size_t)kidx*4096 + didx*64 + f] : be2[didx*64 + f];
    Bf[(size_t)idx*16 + j] = (__bf16)v;
  }
}

// Bcat: cols 0..63 = root (h@root), cols 64..255 = Whh^T (h@Whh.T). K=64.
__global__ void k_pack_Bcat(const float* __restrict__ root, const float* __restrict__ Whh,
                            __bf16* __restrict__ Bf) {
  int idx = blockIdx.x * blockDim.x + threadIdx.x;      // 16*2*32
  if (idx >= 16*2*32) return;
  int lane = idx & 31, s = (idx >> 5) & 1, nt = idx / 64;
  int n = lane & 15, half = lane >> 4;
  int col = nt*16 + n;
  #pragma unroll
  for (int j = 0; j < 16; ++j) {
    int k = s*32 + a_kk(j, half);
    float v = (col < 64) ? root[k*64 + col] : Whh[(col-64)*64 + k];
    Bf[(size_t)idx*16 + j] = (__bf16)v;
  }
}

// Bwih: m @ Wih^T, N=192, K=64.
__global__ void k_pack_Bwih(const float* __restrict__ Wih, __bf16* __restrict__ Bf) {
  int idx = blockIdx.x * blockDim.x + threadIdx.x;      // 12*2*32
  if (idx >= 12*2*32) return;
  int lane = idx & 31, s = (idx >> 5) & 1, nt = idx / 64;
  int n = lane & 15, half = lane >> 4;
  int col = nt*16 + n;
  #pragma unroll
  for (int j = 0; j < 16; ++j) {
    int k = s*32 + a_kk(j, half);
    Bf[(size_t)idx*16 + j] = (__bf16)Wih[col*64 + k];
  }
}

// ----------------------- edge message GEMM (WMMA) ---------------------------
// One wave per 16-edge tile, full N=64 (4 n-tiles). 130 K-slabs.
// Per slab s: A(m, kk) = he[m, s>>1] * t[m, (s&1)*32 + kk]; t hoisted into
// registers for both parities; he==1 bias block lives in column 64 of sh_he.
// B fragments TDM-streamed into a double-buffered LDS chunk shared by all
// 8 waves of the block; wave 0 drives the TDM and TENSORcnt waits.
__global__ __launch_bounds__(256) void k_edge_msg(
    const float* __restrict__ h, const float* __restrict__ he,
    const __bf16* __restrict__ BfE, const int* __restrict__ src,
    const int* __restrict__ dst, float* __restrict__ agg) {
  __shared__ float  sh_he[8][16][65];                   // col 64 == 1.0f
  __shared__ float  sh_t [8][16][65];
  __shared__ __bf16 sB[2][NT_EDGE][CH*512];             // 2 x 40 KB
  int wave = threadIdx.x >> 5, lane = threadIdx.x & 31;
  int tile = blockIdx.x * 8 + wave;                     // 1024 tiles total
  int ebase = tile * 16;
  for (int i = lane; i < 16*64; i += 32) {
    int r = i >> 6, c = i & 63;
    sh_he[wave][r][c] = he[(size_t)(ebase + r)*64 + c];
    sh_t [wave][r][c] = h [(size_t)src[ebase + r]*64 + c];
  }
  if (lane < 16) sh_he[wave][lane][64] = 1.0f;          // bias block (k==64)
  if (threadIdx.x < 32) {
    tdm_load_2d(lds_off(&sB[0][0][0]), BfE,
                CH*256, NT_EDGE, SLABS_E*256);          // chunk 0
    __builtin_amdgcn_s_wait_tensorcnt(0);
  }
  __syncthreads();

  int m = lane & 15, half = lane >> 4;
  // Hoist t into registers: didx = p*32 + half*8 + j (j<8) / p*32+16+half*8+j.
  float t00[8], t01[8], t10[8], t11[8];
  #pragma unroll
  for (int j = 0; j < 8; ++j) {
    t00[j] = sh_t[wave][m][     half*8 + j];
    t01[j] = sh_t[wave][m][16 + half*8 + j];
    t10[j] = sh_t[wave][m][32 + half*8 + j];
    t11[j] = sh_t[wave][m][48 + half*8 + j];
  }

  v8f acc0 = {}, acc1 = {}, acc2 = {}, acc3 = {};
  for (int c = 0; c < NCH; ++c) {
    int buf = c & 1;
    if (threadIdx.x < 32 && c + 1 < NCH)
      tdm_load_2d(lds_off(&sB[buf ^ 1][0][0]), BfE + (size_t)(c+1)*CH*512,
                  CH*256, NT_EDGE, SLABS_E*256);        // prefetch chunk c+1
    #pragma unroll
    for (int sp = 0; sp < CH/2; ++sp) {                 // slab pairs share hv
      float hv = sh_he[wave][m][c*(CH/2) + sp];         // kidx = s>>1
      #pragma unroll
      for (int par = 0; par < 2; ++par) {               // slab parity
        int sl = sp*2 + par;
        v16bf a;
        #pragma unroll
        for (int j = 0; j < 8; ++j) {
          a[j]     = (__bf16)(hv * (par ? t10[j] : t00[j]));
          a[j + 8] = (__bf16)(hv * (par ? t11[j] : t01[j]));
        }
        int bo = sl*512 + lane*16;
        v16bf b0 = *(const v16bf*)&sB[buf][0][bo];
        v16bf b1 = *(const v16bf*)&sB[buf][1][bo];
        v16bf b2 = *(const v16bf*)&sB[buf][2][bo];
        v16bf b3 = *(const v16bf*)&sB[buf][3][bo];
        acc0 = wmma_bf16(a, b0, acc0);
        acc1 = wmma_bf16(a, b1, acc1);
        acc2 = wmma_bf16(a, b2, acc2);
        acc3 = wmma_bf16(a, b3, acc3);
      }
    }
    if (threadIdx.x < 32)
      __builtin_amdgcn_s_wait_tensorcnt(0);             // chunk c+1 landed
    __syncthreads();                                    // publish + reuse-safe
  }
  // C layout: lane -> N = lane%16, VGPR r -> M = half*8 + r. Scatter-add.
  int n = lane & 15;
  #pragma unroll
  for (int r = 0; r < 8; ++r) {
    int er = ebase + half*8 + r;
    size_t base = (size_t)dst[er] * 64;
    atomAddF(&agg[base +      n], acc0[r]);
    atomAddF(&agg[base + 16 + n], acc1[r]);
    atomAddF(&agg[base + 32 + n], acc2[r]);
    atomAddF(&agg[base + 48 + n], acc3[r]);
  }
}

// ------------------------ generic K=64 WMMA GEMM ----------------------------
// C[M x (NT*16)] = A[M x 64] @ Bfrag.  One wave per 16-row m-tile.
// Packed B (<= 32 KB) is TDM-loaded into LDS once, shared by all waves.
__global__ __launch_bounds__(256) void k_gemm_k64(
    const float* __restrict__ A, const __bf16* __restrict__ Bfrag,
    float* __restrict__ C, int NT, int ldc) {
  __shared__ float  sA[8][16][65];
  __shared__ __bf16 sB[16*2*32*16];                     // up to 32 KB
  int wave = threadIdx.x >> 5, lane = threadIdx.x & 31;
  int mt = blockIdx.x * 8 + wave;
  int rbase = mt * 16;
  for (int i = lane; i < 16*64; i += 32)
    sA[wave][i >> 6][i & 63] = A[(size_t)(rbase + (i >> 6))*64 + (i & 63)];
  if (threadIdx.x < 32) {
    tdm_load_2d(lds_off(&sB[0]), Bfrag, (unsigned)NT*512, 1, (unsigned)NT*512);
    __builtin_amdgcn_s_wait_tensorcnt(0);
  }
  __syncthreads();

  int mrow = lane & 15, half = lane >> 4, n = lane & 15;
  v16bf a0, a1;
  #pragma unroll
  for (int j = 0; j < 16; ++j) {
    int kk = a_kk(j, half);
    a0[j] = (__bf16)sA[wave][mrow][kk];
    a1[j] = (__bf16)sA[wave][mrow][kk + 32];
  }
  for (int nt = 0; nt < NT; ++nt) {
    const __bf16* bp = &sB[((size_t)nt*64 + lane)*16];  // [nt][s][lane][16]
    v16bf b0 = *(const v16bf*)bp;
    v16bf b1 = *(const v16bf*)(bp + 32*16);
    v8f c = {};
    c = wmma_bf16(a0, b0, c);
    c = wmma_bf16(a1, b1, c);
    #pragma unroll
    for (int r = 0; r < 8; ++r)
      C[(size_t)(rbase + half*8 + r)*ldc + nt*16 + n] = c[r];
  }
}

// --------------------------- per-iteration pointwise ------------------------
__global__ void k_mkernel(const float* __restrict__ agg, const float* __restrict__ inv,
                          const float* __restrict__ G0, const float* __restrict__ conv_b,
                          float* __restrict__ m) {
  int i = blockIdx.x * blockDim.x + threadIdx.x;        // NNODES*64
  int nd = i >> 6, f = i & 63;
  m[i] = lrelu_f(agg[i]*inv[nd] + G0[(size_t)nd*256 + f] + conv_b[f]);
}

__global__ void k_gru(const float* __restrict__ Gi, const float* __restrict__ G0,
                      const float* __restrict__ bih, const float* __restrict__ bhh,
                      float* __restrict__ h) {
  int i = blockIdx.x * blockDim.x + threadIdx.x;        // NNODES*64
  int nd = i >> 6, f = i & 63;
  float gi_r = Gi[(size_t)nd*192 +       f] + bih[f];
  float gi_z = Gi[(size_t)nd*192 +  64 + f] + bih[64 + f];
  float gi_n = Gi[(size_t)nd*192 + 128 + f] + bih[128 + f];
  float gh_r = G0[(size_t)nd*256 +  64 + f] + bhh[f];
  float gh_z = G0[(size_t)nd*256 + 128 + f] + bhh[64 + f];
  float gh_n = G0[(size_t)nd*256 + 192 + f] + bhh[128 + f];
  float r = sigm_f(gi_r + gh_r);
  float z = sigm_f(gi_z + gh_z);
  float nn = tanhf(gi_n + r * gh_n);
  h[i] = (1.f - z) * nn + z * h[i];
}

// ------------------------------- Set2Set head -------------------------------
// q_star = h_lstm = c_lstm = 0 -> gates depend only on biases; q identical
// for every graph.
__global__ void k_qvec(const float* __restrict__ lbih, const float* __restrict__ lbhh,
                       float* __restrict__ q) {
  int f = threadIdx.x; if (f >= 64) return;
  float gi = lbih[f]       + lbhh[f];
  float gg = lbih[128 + f] + lbhh[128 + f];
  float go = lbih[192 + f] + lbhh[192 + f];
  float cl = sigm_f(gi) * tanhf(gg);                    // c_prev == 0
  q[f] = sigm_f(go) * tanhf(cl);
}

__global__ void k_escore(const float* __restrict__ h, const float* __restrict__ q,
                         float* __restrict__ e) {
  int nd = blockIdx.x * blockDim.x + threadIdx.x;
  if (nd >= NNODES) return;
  float s = 0.f;
  #pragma unroll 8
  for (int f = 0; f < 64; ++f) s += h[(size_t)nd*64 + f] * q[f];
  e[nd] = s;
}

__global__ __launch_bounds__(256) void k_pool(const float* __restrict__ e,
                                              const int* __restrict__ batch,
                                              const float* __restrict__ h,
                                              float* __restrict__ rpool) {
  int g = blockIdx.x, tid = threadIdx.x;
  __shared__ float red[256];
  __shared__ float rp[4][64];
  float mx = -3.4e38f;
  for (int nd = tid; nd < NNODES; nd += 256)
    if (batch[nd] == g) mx = fmaxf(mx, e[nd]);
  red[tid] = mx; __syncthreads();
  for (int s = 128; s > 0; s >>= 1) {
    if (tid < s) red[tid] = fmaxf(red[tid], red[tid + s]);
    __syncthreads();
  }
  float emax = red[0]; __syncthreads();
  float sum = 0.f;
  for (int nd = tid; nd < NNODES; nd += 256)
    if (batch[nd] == g) sum += __expf(e[nd] - emax);
  red[tid] = sum; __syncthreads();
  for (int s = 128; s > 0; s >>= 1) {
    if (tid < s) red[tid] += red[tid + s];
    __syncthreads();
  }
  float inva = 1.0f / red[0]; __syncthreads();
  int f = tid & 63, grp = tid >> 6;
  float acc = 0.f;
  for (int nd = grp; nd < NNODES; nd += 4)
    if (batch[nd] == g) acc += __expf(e[nd] - emax) * inva * h[(size_t)nd*64 + f];
  rp[grp][f] = acc; __syncthreads();
  if (grp == 0) rpool[g*64 + f] = rp[0][f] + rp[1][f] + rp[2][f] + rp[3][f];
}

__global__ void k_final(const float* __restrict__ q, const float* __restrict__ rpool,
                        const float* __restrict__ Wout, const float* __restrict__ bout,
                        float* __restrict__ out) {
  int t = threadIdx.x; if (t >= 32) return;
  int g = t >> 1, c = t & 1;
  float acc = bout[c];
  for (int f = 0; f < 64; ++f) acc += q[f] * Wout[f*2 + c];
  for (int f = 0; f < 64; ++f) acc += rpool[g*64 + f] * Wout[(64 + f)*2 + c];
  out[g*2 + c] = acc;
}

// ------------------------------- host side ----------------------------------
static inline char* carve(char*& p, size_t bytes) {
  char* r = p;
  p += (bytes + 255) & ~(size_t)255;
  return r;
}

extern "C" void kernel_launch(void* const* d_in, const int* in_sizes, int n_in,
                              void* d_out, int out_size, void* d_ws, size_t ws_size,
                              hipStream_t stream) {
  const float* x         = (const float*)d_in[0];
  const float* edge_attr = (const float*)d_in[1];
  const int*   eidx      = (const int*)  d_in[2];
  const int*   batch     = (const int*)  d_in[3];
  const float* W0     = (const float*)d_in[4];
  const float* b0     = (const float*)d_in[5];
  const float* We1    = (const float*)d_in[6];
  const float* be1    = (const float*)d_in[7];
  const float* We2    = (const float*)d_in[8];
  const float* be2    = (const float*)d_in[9];
  const float* root   = (const float*)d_in[10];
  const float* conv_b = (const float*)d_in[11];
  const float* Wih    = (const float*)d_in[12];
  const float* Whh    = (const float*)d_in[13];
  const float* bih    = (const float*)d_in[14];
  const float* bhh    = (const float*)d_in[15];
  const float* lbih   = (const float*)d_in[18];
  const float* lbhh   = (const float*)d_in[19];
  const float* Wout   = (const float*)d_in[20];
  const float* bout   = (const float*)d_in[21];
  const int* src = eidx;
  const int* dst = eidx + NEDGES;

  char* p = (char*)d_ws;                      // needs ~15.3 MB of workspace
  float* h      = (float*)carve(p, (size_t)NNODES*64*4);
  float* agg    = (float*)carve(p, (size_t)NNODES*64*4);
  float* G0     = (float*)carve(p, (size_t)NNODES*256*4);
  float* mbuf   = (float*)carve(p, (size_t)NNODES*64*4);
  float* Gi     = (float*)carve(p, (size_t)NNODES*192*4);
  float* hebuf  = (float*)carve(p, (size_t)NEDGES*64*4);
  float* deg    = (float*)carve(p, (size_t)NNODES*4);
  float* invdeg = (float*)carve(p, (size_t)NNODES*4);
  __bf16* BfE   = (__bf16*)carve(p, (size_t)NT_EDGE*SLABS_E*32*16*2);
  __bf16* Bfcat = (__bf16*)carve(p, (size_t)16*2*32*16*2);
  __bf16* Bfwih = (__bf16*)carve(p, (size_t)12*2*32*16*2);
  float* qv     = (float*)carve(p, 64*4);
  float* ebuf   = (float*)carve(p, (size_t)NNODES*4);
  float* rpool  = (float*)carve(p, (size_t)BGRAPH*64*4);

  // one-time stages
  k_node_init<<<NNODES*64/256, 256, 0, stream>>>(x, W0, b0, h);
  k_edge_mlp <<<NEDGES*64/256, 256, 0, stream>>>(edge_attr, We1, be1, hebuf);
  k_zero     <<<NNODES/256, 256, 0, stream>>>(deg, NNODES);
  k_deg      <<<NEDGES/256, 256, 0, stream>>>(dst, deg);
  k_invdeg   <<<NNODES/256, 256, 0, stream>>>(deg, invdeg);
  k_pack_edgeB<<<(NT_EDGE*SLABS_E*32 + 255)/256, 256, 0, stream>>>(We2, be2, BfE);
  k_pack_Bcat <<<(16*2*32 + 255)/256, 256, 0, stream>>>(root, Whh, Bfcat);
  k_pack_Bwih <<<(12*2*32 + 255)/256, 256, 0, stream>>>(Wih, Bfwih);

  // 6 message-passing iterations
  for (int it = 0; it < 6; ++it) {
    k_zero    <<<NNODES*64/256, 256, 0, stream>>>(agg, NNODES*64);
    k_edge_msg<<<NEDGES/16/8, 256, 0, stream>>>(h, hebuf, BfE, src, dst, agg);
    k_gemm_k64<<<NNODES/16/8, 256, 0, stream>>>(h, Bfcat, G0, 16, 256);
    k_mkernel <<<NNODES*64/256, 256, 0, stream>>>(agg, invdeg, G0, conv_b, mbuf);
    k_gemm_k64<<<NNODES/16/8, 256, 0, stream>>>(mbuf, Bfwih, Gi, 12, 192);
    k_gru     <<<NNODES*64/256, 256, 0, stream>>>(Gi, G0, bih, bhh, h);
  }

  // Set2Set (1 step) + output head
  k_qvec  <<<1, 64, 0, stream>>>(lbih, lbhh, qv);
  k_escore<<<NNODES/256, 256, 0, stream>>>(h, qv, ebuf);
  k_pool  <<<BGRAPH, 256, 0, stream>>>(ebuf, batch, h, rpool);
  k_final <<<1, 32, 0, stream>>>(qv, rpool, Wout, bout, (float*)d_out);
}
